// PCAutoEncoder_19731079758278
// MI455X (gfx1250) — matrix operand order repaired and checked by reading, not compiled, not run on hardware
//
#include <hip/hip_runtime.h>
#include <hip/hip_bf16.h>

#define BATCH 8
#define NPTS  16384

typedef __attribute__((ext_vector_type(16))) _Float16 v16h;
typedef __attribute__((ext_vector_type(8)))  _Float16 v8h;
typedef __attribute__((ext_vector_type(8)))  float    v8f;
typedef __attribute__((ext_vector_type(4)))  unsigned int uint32x4;
typedef __attribute__((ext_vector_type(8)))  int      int32x8;
typedef __attribute__((ext_vector_type(4)))  int      int32x4;

// ---------------------------------------------------------------------------
// WMMA GEMM:  Y[O x M] = W[O x Kp] * X[M x Kp]^T   (f16 in, f32 acc, f16 out)
// X is stored transposed (M rows of Kp halfs) so every fragment slice is a
// contiguous 32-byte run. The 64x32 X tile is staged into LDS by the Tensor
// Data Mover (tensor_load_to_lds + s_wait_tensorcnt), then consumed by
// v_wmma_f32_16x16x32_f16. Block = 256 thr (8 waves), tile 64(O) x 64(M).
// ---------------------------------------------------------------------------
__global__ void wmma_gemm_f16(const _Float16* __restrict__ W,  // O x Kp
                              const _Float16* __restrict__ X,  // M x Kp
                              _Float16* __restrict__ Y,        // O x M
                              int O, int Kp, int M) {
  __shared__ _Float16 sX[64 * 32];  // 64 m-rows x 32 k halfs = 4 KB

  const int tid  = threadIdx.x;
  const int lane = tid & 31;
  const int wave = tid >> 5;
  const int half = lane >> 4;  // 0: lanes 0-15, 1: lanes 16-31
  const int l16  = lane & 15;

  const int rowBase  = blockIdx.y * 64 + (wave & 3) * 16;
  const int colBlock = blockIdx.x * 64;
  const int colLocal = (wave >> 2) * 32;

  const unsigned ldsAddr = (unsigned)(size_t)(&sX[0]);

  v8f acc0 = {}, acc1 = {};

  for (int kb = 0; kb < Kp; kb += 32) {
#if defined(__has_builtin) && __has_builtin(__builtin_amdgcn_tensor_load_to_lds)
    // ---- TDM: stage 64 rows x 32 halfs of X into LDS (one instruction) ----
    if (tid == 0) {
      unsigned long long gaddr =
          (unsigned long long)(size_t)X +
          ((unsigned long long)(unsigned)colBlock * (unsigned)Kp +
           (unsigned)kb) * 2ull;
      uint32x4 g0;
      g0.x = 0x1u;      // count=1, user descriptor, no gather
      g0.y = ldsAddr;   // LDS destination byte address
      g0.z = (unsigned)(gaddr & 0xFFFFFFFFull);
      g0.w = (unsigned)((gaddr >> 32) & 0x1FFFFFFull) | (2u << 30);  // type=2
      unsigned td0 = (unsigned)Kp;  // tensor_dim0 (k extent)
      unsigned td1 = (unsigned)M;   // tensor_dim1 (m extent)
      unsigned long long st0 = (unsigned)Kp;  // dim0 stride (elements)
      int32x8 g1;
      g1[0] = (int)(1u << 16);                          // data_size = 2 bytes
      g1[1] = (int)((td0 & 0xFFFFu) << 16);             // tensor_dim0 lo16
      g1[2] = (int)(((td0 >> 16) & 0xFFFFu) | ((td1 & 0xFFFFu) << 16));
      g1[3] = (int)(((td1 >> 16) & 0xFFFFu) | (32u << 16));  // tile_dim0 = 32
      g1[4] = (int)64;                                  // tile_dim1 = 64
      g1[5] = (int)(st0 & 0xFFFFFFFFull);               // dim0_stride lo32
      g1[6] = (int)((st0 >> 32) & 0xFFFFull);           // dim0_stride hi16
      g1[7] = 0;
      int32x4 z4 = {0, 0, 0, 0};
      int32x8 z8 = {0, 0, 0, 0, 0, 0, 0, 0};
      // clang-23 / therock-10.0 toolchain: 6-arg form
      __builtin_amdgcn_tensor_load_to_lds(g0, g1, z4, z4, z8, 0);
    }
#if __has_builtin(__builtin_amdgcn_s_wait_tensorcnt)
    if (wave == 0) __builtin_amdgcn_s_wait_tensorcnt(0);
#endif
#else
    // fallback: cooperative 32-byte vector copies
    for (int t = tid; t < 64 * 2; t += 256) {
      int r = t >> 1, c = t & 1;
      *(v16h*)(sX + r * 32 + c * 16) =
          *(const v16h*)(X + (size_t)(colBlock + r) * Kp + kb + c * 16);
    }
#endif
    __syncthreads();

    if (kb + 32 < Kp)
      __builtin_prefetch(W + (size_t)(rowBase + l16) * Kp + kb + 32, 0, 1);

    // A fragment: one contiguous 32-byte run of the lane's W row (k = 16*half + s)
    const _Float16* wr = W + (size_t)(rowBase + l16) * Kp + kb + half * 16;
    v8h alo = *(const v8h*)(wr);
    v8h ahi = *(const v8h*)(wr + 8);
    v16h a  = __builtin_shufflevector(alo, ahi, 0, 1, 2, 3, 4, 5, 6, 7,
                                      8, 9, 10, 11, 12, 13, 14, 15);

    // B fragments: contiguous 32-byte LDS reads (lane = m row, K-major)
    const _Float16* x0 = sX + (size_t)(colLocal + l16) * 32 + half * 16;
    const _Float16* x1 = sX + (size_t)(colLocal + 16 + l16) * 32 + half * 16;
    v16h b0 = *(const v16h*)x0;
    v16h b1 = *(const v16h*)x1;

    acc0 = __builtin_amdgcn_wmma_f32_16x16x32_f16(false, a, false, b0,
                                                  (short)0, acc0, false, false);
    acc1 = __builtin_amdgcn_wmma_f32_16x16x32_f16(false, a, false, b1,
                                                  (short)0, acc1, false, false);
    __syncthreads();  // protect sX before next TDM overwrite
  }

  // C/D layout: lane = N col, vgpr v -> row v (lanes 0-15) / v+8 (lanes 16-31)
#pragma unroll
  for (int v = 0; v < 8; ++v) {
    size_t r = (size_t)(rowBase + v + 8 * half) * M;
    Y[r + colBlock + colLocal + l16]      = (_Float16)acc0[v];
    Y[r + colBlock + colLocal + 16 + l16] = (_Float16)acc1[v];
  }
}

// ---------------------------------------------------------------------------
// Farthest point sampling: one workgroup (256 thr) per batch, LDS argmax tree
// ---------------------------------------------------------------------------
__global__ void fps_kernel(const float* __restrict__ xyz, int Pn, int npoint,
                           int* __restrict__ inds, float* __restrict__ dmin) {
  const int b   = blockIdx.x;
  const int tid = threadIdx.x;
  const float* p = xyz + (size_t)b * Pn * 3;
  float* d = dmin + (size_t)b * Pn;
  int* out = inds + (size_t)b * npoint;

  __shared__ float sval[256];
  __shared__ int   sidx[256];
  __shared__ int   sLast;

  for (int i = tid; i < Pn; i += 256) d[i] = 1e10f;
  __syncthreads();

  int last = 0;
  for (int s = 0; s < npoint; ++s) {
    if (tid == 0) out[s] = last;
    float cx = p[last * 3], cy = p[last * 3 + 1], cz = p[last * 3 + 2];
    float best = -1.0f;
    int   bi   = 0;
    for (int i = tid; i < Pn; i += 256) {
      float dx = p[i * 3] - cx, dy = p[i * 3 + 1] - cy, dz = p[i * 3 + 2] - cz;
      float dist = dx * dx + dy * dy + dz * dz;
      float dm = fminf(d[i], dist);
      d[i] = dm;
      if (dm > best) { best = dm; bi = i; }
    }
    sval[tid] = best;
    sidx[tid] = bi;
    __syncthreads();
    for (int off = 128; off > 0; off >>= 1) {
      if (tid < off && sval[tid + off] > sval[tid]) {
        sval[tid] = sval[tid + off];
        sidx[tid] = sidx[tid + off];
      }
      __syncthreads();
    }
    if (tid == 0) sLast = sidx[0];
    __syncthreads();
    last = sLast;
  }
}

__global__ void gather_xyz_kernel(const float* __restrict__ xyz,
                                  const int* __restrict__ inds,
                                  float* __restrict__ new_xyz, int Pn, int S) {
  int g = blockIdx.x * blockDim.x + threadIdx.x;
  if (g >= BATCH * S) return;
  int b = g / S, s = g % S;
  int j = inds[b * S + s];
  const float* src = xyz + ((size_t)b * Pn + j) * 3;
  float* dst = new_xyz + ((size_t)b * S + s) * 3;
  dst[0] = src[0]; dst[1] = src[1]; dst[2] = src[2];
}

// ---------------------------------------------------------------------------
// Ball query: one wave per (b,s); ordered first-nsample via wave32 ballot
// ---------------------------------------------------------------------------
__global__ void ball_query_kernel(const float* __restrict__ new_xyz,
                                  const float* __restrict__ xyz,
                                  int Pn, int S, float r2, int ns,
                                  int* __restrict__ idx) {
  const int b = blockIdx.y, s = blockIdx.x;
  const int lane = threadIdx.x;
  const float* p = xyz + (size_t)b * Pn * 3;
  const float* c = new_xyz + ((size_t)b * S + s) * 3;
  int* o = idx + ((size_t)b * S + s) * ns;
  float cx = c[0], cy = c[1], cz = c[2];

  int count = 0;
  for (int base = 0; base < Pn && count < ns; base += 32) {
    int i = base + lane;
    bool hit = false;
    if (i < Pn) {
      float dx = p[i * 3] - cx, dy = p[i * 3 + 1] - cy, dz = p[i * 3 + 2] - cz;
      hit = (dx * dx + dy * dy + dz * dz) < r2;
    }
    unsigned long long m = __ballot(hit);
    int prefix = __popcll(m & ((1ull << lane) - 1ull));
    if (hit && (count + prefix) < ns) o[count + prefix] = i;
    count += __popcll(m);
  }
  if (count == 0) {
    if (lane == 0) o[0] = 0;
    count = 1;
  }
  int first = o[0];
  for (int j = count + lane; j < ns; j += 32) o[j] = first;
}

// ---------------------------------------------------------------------------
// SA grouping -> im2col f16 matrix X[M x Kp] (row per sample, contiguous)
// cols 0..2 = (xyz[j]-center)/radius, cols 3..3+Cf-1 = gathered feats, pad 0
// ---------------------------------------------------------------------------
__global__ void sa_group_kernel(const float* __restrict__ xyz,
                                const float* __restrict__ new_xyz,
                                const float* __restrict__ feats,
                                const int* __restrict__ idx,
                                _Float16* __restrict__ X,
                                int Pn, int S, int ns, int Cf, int Kp,
                                float inv_r) {
  int m = blockIdx.x * blockDim.x + threadIdx.x;
  int M = BATCH * S * ns;
  if (m >= M) return;
  int n  = m % ns;
  int bs = m / ns;
  int b  = bs / S;
  int j  = idx[(size_t)bs * ns + n];
  const float* pj = xyz + ((size_t)b * Pn + j) * 3;
  const float* c  = new_xyz + (size_t)bs * 3;
  _Float16* row = X + (size_t)m * Kp;
  row[0] = (_Float16)((pj[0] - c[0]) * inv_r);
  row[1] = (_Float16)((pj[1] - c[1]) * inv_r);
  row[2] = (_Float16)((pj[2] - c[2]) * inv_r);
  for (int cc = 0; cc < Cf; ++cc)
    row[3 + cc] = (_Float16)feats[((size_t)b * Cf + cc) * Pn + j];
  for (int k = 3 + Cf; k < Kp; ++k) row[k] = (_Float16)0.f;
}

// ---------------------------------------------------------------------------
// Batch-norm statistics (batch stats, matching reference), apply + ReLU.
// bn_relu_t writes the TRANSPOSED (M x Kout) layout feeding the next GEMM.
// ---------------------------------------------------------------------------
__global__ void bn_stats_kernel(const _Float16* __restrict__ Y, int M,
                                float* __restrict__ mean,
                                float* __restrict__ var) {
  int o = blockIdx.x, tid = threadIdx.x;
  const _Float16* row = Y + (size_t)o * M;
  float s1 = 0.f, s2 = 0.f;
  for (int i = tid; i < M; i += 256) {
    float x = (float)row[i];
    s1 += x;
    s2 += x * x;
  }
  __shared__ float a1[256], a2[256];
  a1[tid] = s1; a2[tid] = s2;
  __syncthreads();
  for (int off = 128; off > 0; off >>= 1) {
    if (tid < off) { a1[tid] += a1[tid + off]; a2[tid] += a2[tid + off]; }
    __syncthreads();
  }
  if (tid == 0) {
    float mu = a1[0] / (float)M;
    mean[o] = mu;
    var[o]  = a2[0] / (float)M - mu * mu;
  }
}

__global__ void bn_relu_t_kernel(const _Float16* __restrict__ Y,  // O x M
                                 const float* __restrict__ mean,
                                 const float* __restrict__ var,
                                 const float* __restrict__ gamma,
                                 const float* __restrict__ beta,
                                 _Float16* __restrict__ dst,      // M x Kout
                                 int O, int Kout, int M) {
  int g = blockIdx.x * blockDim.x + threadIdx.x;
  if (g >= Kout * M) return;
  int o = g % Kout, m = g / Kout;
  float val = 0.f;
  if (o < O) {
    float x = (float)Y[(size_t)o * M + m];
    val = (x - mean[o]) * rsqrtf(var[o] + 1e-5f) * gamma[o] + beta[o];
    val = fmaxf(val, 0.f);
  }
  dst[g] = (_Float16)val;
}

__global__ void maxpool_kernel(const _Float16* __restrict__ act,  // M x O
                               float* __restrict__ feats,
                               int O, int S, int ns) {
  int g = blockIdx.x * blockDim.x + threadIdx.x;
  if (g >= BATCH * O * S) return;
  int s = g % S;
  int t = g / S;
  int o = t % O;
  int b = t / O;
  const _Float16* base = act + (size_t)(b * S + s) * ns * O + o;
  float mx = -1e30f;
  for (int n = 0; n < ns; ++n) mx = fmaxf(mx, (float)base[(size_t)n * O]);
  feats[g] = mx;  // layout (b*O + o)*S + s
}

// ---------------------------------------------------------------------------
// FP: 3-NN inverse-distance interpolation + concat -> X[M x Kp], M = B*n1
// ---------------------------------------------------------------------------
__global__ void fp_group_kernel(const float* __restrict__ xyz1,
                                const float* __restrict__ xyz2,
                                const float* __restrict__ feats1,
                                const float* __restrict__ feats2,
                                _Float16* __restrict__ X,
                                int n1, int n2, int C1, int C2, int Kp) {
  int g = blockIdx.x * blockDim.x + threadIdx.x;
  int M = BATCH * n1;
  if (g >= M) return;
  int i = g % n1, b = g / n1;
  const float* q   = xyz1 + ((size_t)b * n1 + i) * 3;
  const float* pts = xyz2 + (size_t)b * n2 * 3;

  float bd0 = 1e30f, bd1 = 1e30f, bd2 = 1e30f;
  int   bi0 = 0, bi1 = 0, bi2 = 0;
  for (int j = 0; j < n2; ++j) {
    float dx = pts[j * 3] - q[0], dy = pts[j * 3 + 1] - q[1],
          dz = pts[j * 3 + 2] - q[2];
    float d2 = dx * dx + dy * dy + dz * dz;
    if (d2 < bd0)      { bd2=bd1; bi2=bi1; bd1=bd0; bi1=bi0; bd0=d2; bi0=j; }
    else if (d2 < bd1) { bd2=bd1; bi2=bi1; bd1=d2; bi1=j; }
    else if (d2 < bd2) { bd2=d2; bi2=j; }
  }
  float w0 = 1.f / (sqrtf(fmaxf(bd0, 0.f)) + 1e-8f);
  float w1 = 1.f / (sqrtf(fmaxf(bd1, 0.f)) + 1e-8f);
  float w2 = 1.f / (sqrtf(fmaxf(bd2, 0.f)) + 1e-8f);
  float ws = w0 + w1 + w2;
  w0 /= ws; w1 /= ws; w2 /= ws;

  _Float16* row = X + (size_t)g * Kp;
  int k = 0;
  for (int c = 0; c < C1; ++c)
    row[k++] = (_Float16)feats1[((size_t)b * C1 + c) * n1 + i];
  const float* f2b = feats2 + (size_t)b * C2 * n2;
  for (int c = 0; c < C2; ++c) {
    float v = w0 * f2b[c * n2 + bi0] + w1 * f2b[c * n2 + bi1] +
              w2 * f2b[c * n2 + bi2];
    row[k++] = (_Float16)v;
  }
  for (; k < Kp; ++k) row[k] = (_Float16)0.f;
}

__global__ void act_to_feats_kernel(const _Float16* __restrict__ act,  // M x C
                                    float* __restrict__ feats, int C, int n) {
  int g = blockIdx.x * blockDim.x + threadIdx.x;
  if (g >= BATCH * C * n) return;
  int i = g % n;
  int t = g / n;
  int c = t % C;
  int b = t / C;
  feats[g] = (float)act[((size_t)b * n + i) * C + c];
}

__global__ void final_out_kernel(const _Float16* __restrict__ act,  // M x C
                                 float* __restrict__ out, int total) {
  int g = blockIdx.x * blockDim.x + threadIdx.x;
  if (g >= total) return;
  out[g] = (float)act[g];  // layouts coincide: (b, i, c)
}

__global__ void prep_w_kernel(const float* __restrict__ W,
                              _Float16* __restrict__ Wp, int O, int C, int Kp) {
  int g = blockIdx.x * blockDim.x + threadIdx.x;
  if (g >= O * Kp) return;
  int o = g / Kp, k = g % Kp;
  Wp[g] = (k < C) ? (_Float16)W[o * C + k] : (_Float16)0.f;
}

// ---------------------------------------------------------------------------
// Host orchestration
// ---------------------------------------------------------------------------
struct LayerP { const float *W, *g, *b; int O, C; };

extern "C" void kernel_launch(void* const* d_in, const int* in_sizes, int n_in,
                              void* d_out, int out_size, void* d_ws,
                              size_t ws_size, hipStream_t stream) {
  (void)in_sizes; (void)n_in; (void)out_size; (void)ws_size;

  int pi = 0;
  const float* pc = (const float*)d_in[pi++];  // (B, P, 3)
  auto getL = [&](int O, int C) {
    LayerP L;
    L.W = (const float*)d_in[pi++];
    L.g = (const float*)d_in[pi++];
    L.b = (const float*)d_in[pi++];
    L.O = O; L.C = C;
    return L;
  };
  LayerP sa1[3] = { getL(64, 3),    getL(64, 64),   getL(128, 64) };
  LayerP sa2[3] = { getL(128, 131), getL(128, 128), getL(256, 128) };
  LayerP sa3[3] = { getL(128, 259), getL(128, 128), getL(256, 128) };
  LayerP fp2[2] = { getL(256, 512), getL(256, 256) };
  LayerP fp1[2] = { getL(256, 384), getL(256, 256) };
  LayerP fp0[2] = { getL(256, 256), getL(256, 256) };

  // workspace bump allocator
  char*  ws  = (char*)d_ws;
  size_t off = 0;
  auto alloc = [&](size_t bytes) -> void* {
    void* p = ws + off;
    off = (off + bytes + 255) & ~(size_t)255;
    return p;
  };

  float*     dmin    = (float*)alloc((size_t)BATCH * NPTS * 4);
  int*       indsBuf = (int*)alloc((size_t)BATCH * 1024 * 4);
  int*       idxBuf  = (int*)alloc((size_t)BATCH * 1024 * 64 * 4);
  float*     xyz1    = (float*)alloc((size_t)BATCH * 1024 * 3 * 4);
  float*     xyz2    = (float*)alloc((size_t)BATCH * 512 * 3 * 4);
  float*     xyz3    = (float*)alloc((size_t)BATCH * 256 * 3 * 4);
  float*     f1      = (float*)alloc((size_t)BATCH * 128 * 1024 * 4);
  float*     f2      = (float*)alloc((size_t)BATCH * 256 * 512 * 4);
  float*     f3      = (float*)alloc((size_t)BATCH * 256 * 256 * 4);
  float*     fp2f    = (float*)alloc((size_t)BATCH * 256 * 512 * 4);
  float*     fp1f    = (float*)alloc((size_t)BATCH * 256 * 1024 * 4);
  float*     mean    = (float*)alloc(256 * 4);
  float*     var     = (float*)alloc(256 * 4);
  _Float16*  Wp      = (_Float16*)alloc((size_t)256 * 512 * 2);
  _Float16*  bufX    = (_Float16*)alloc((size_t)128 * 524288 * 2);
  _Float16*  bufY    = (_Float16*)alloc((size_t)128 * 524288 * 2);

  auto run_layers = [&](const LayerP* Ls, int nl, int Kp_first, int M) {
    int Kp = Kp_first;
    for (int l = 0; l < nl; ++l) {
      const LayerP& L = Ls[l];
      int nW = L.O * Kp;
      prep_w_kernel<<<(nW + 255) / 256, 256, 0, stream>>>(L.W, Wp, L.O, L.C, Kp);
      dim3 grid(M / 64, L.O / 64);
      wmma_gemm_f16<<<grid, 256, 0, stream>>>(Wp, bufX, bufY, L.O, Kp, M);
      bn_stats_kernel<<<L.O, 256, 0, stream>>>(bufY, M, mean, var);
      int tot = L.O * M;
      bn_relu_t_kernel<<<(tot + 255) / 256, 256, 0, stream>>>(
          bufY, mean, var, L.g, L.b, bufX, L.O, L.O, M);
      Kp = L.O;  // O is always a multiple of 32
    }
  };

  auto run_sa = [&](const float* xyz, int Pn, const float* feats, int Cf,
                    int S, float radius, int ns, const LayerP* Ls,
                    float* new_xyz, float* out_feats) {
    fps_kernel<<<BATCH, 256, 0, stream>>>(xyz, Pn, S, indsBuf, dmin);
    gather_xyz_kernel<<<(BATCH * S + 255) / 256, 256, 0, stream>>>(
        xyz, indsBuf, new_xyz, Pn, S);
    ball_query_kernel<<<dim3(S, BATCH), 32, 0, stream>>>(
        new_xyz, xyz, Pn, S, radius * radius, ns, idxBuf);
    int Kp0 = ((3 + Cf) + 31) / 32 * 32;
    int M   = BATCH * S * ns;
    sa_group_kernel<<<(M + 255) / 256, 256, 0, stream>>>(
        xyz, new_xyz, feats, idxBuf, bufX, Pn, S, ns, Cf, Kp0, 1.f / radius);
    run_layers(Ls, 3, Kp0, M);
    int Ol = Ls[2].O;
    maxpool_kernel<<<(BATCH * Ol * S + 255) / 256, 256, 0, stream>>>(
        bufX, out_feats, Ol, S, ns);
  };

  auto run_fp = [&](const float* x1, const float* x2, const float* fe1, int C1,
                    const float* fe2, int C2, int n1, int n2, const LayerP* Ls,
                    float* out_feats) {
    int Kp0 = ((C1 + C2) + 31) / 32 * 32;
    int M   = BATCH * n1;
    fp_group_kernel<<<(M + 255) / 256, 256, 0, stream>>>(
        x1, x2, fe1, fe2, bufX, n1, n2, C1, C2, Kp0);
    run_layers(Ls, 2, Kp0, M);
    if (out_feats)
      act_to_feats_kernel<<<(BATCH * Ls[1].O * n1 + 255) / 256, 256, 0,
                            stream>>>(bufX, out_feats, Ls[1].O, n1);
  };

  // --- pipeline ---
  run_sa(pc,   NPTS, nullptr, 0,   1024, 0.1f, 64, sa1, xyz1, f1);
  run_sa(xyz1, 1024, f1,      128, 512,  0.2f, 32, sa2, xyz2, f2);
  run_sa(xyz2, 512,  f2,      256, 256,  0.4f, 16, sa3, xyz3, f3);

  run_fp(xyz2, xyz3, f2, 256, f3,   256, 512,   256,  fp2, fp2f);
  run_fp(xyz1, xyz2, f1, 128, fp2f, 256, 1024,  512,  fp1, fp1f);
  run_fp(pc,   xyz1, nullptr, 0, fp1f, 256, NPTS, 1024, fp0, nullptr);

  final_out_kernel<<<(BATCH * NPTS * 256 + 255) / 256, 256, 0, stream>>>(
      bufX, (float*)d_out, BATCH * NPTS * 256);
}
